// NN_1580547975065
// MI455X (gfx1250) — matrix operand order repaired and checked by reading, not compiled, not run on hardware
//
#include <hip/hip_runtime.h>
#include <math.h>

// ---------------- types ----------------
typedef __attribute__((ext_vector_type(16))) __bf16 v16bf;
typedef __attribute__((ext_vector_type(8)))  float  v8f;

struct U32x8 { uint4 lo, hi; };

// ---------------- helpers ----------------
__device__ __forceinline__ unsigned short f32_to_bf16(float f) {
    unsigned int u = __float_as_uint(f);
    unsigned int r = (u + 0x7FFFu + ((u >> 16) & 1u)) >> 16;   // RNE
    return (unsigned short)r;
}
__device__ __forceinline__ float hsig(float v) {
    return fminf(fmaxf(0.2f * v + 0.5f, 0.0f), 1.0f);
}
__device__ __forceinline__ void atomicMaxF(float* addr, float v) {
    if (v >= 0.0f) atomicMax((int*)addr, __float_as_int(v));
    else           atomicMin((unsigned int*)addr, (unsigned int)__float_as_int(v));
}

// CDNA5 async copy: global -> LDS, 16 bytes per lane, tracked by ASYNCcnt.
__device__ __forceinline__ void async_b128(unsigned lds_off, const void* gptr) {
    asm volatile("global_load_async_to_lds_b128 %0, %1, off"
                 :: "v"(lds_off), "v"(gptr) : "memory");
}
__device__ __forceinline__ void async_wait0() {
    asm volatile("s_wait_asynccnt 0x0" ::: "memory");
}
__device__ __forceinline__ unsigned lds_addr(const void* p) {
    // generic LDS pointer: low 32 bits are the LDS byte offset
    return (unsigned)(unsigned long long)p;
}

// A-fragment (16-bit A 16x32): lane<16 holds K {0..15}, lane>=16 holds {16..31}
__device__ __forceinline__ v16bf load_a_frag(const unsigned short* As, int row, int ksel) {
    U32x8 u;
    u.lo = *(const uint4*)(As + row * 32 + ksel);
    u.hi = *(const uint4*)(As + row * 32 + ksel + 8);
    return __builtin_bit_cast(v16bf, u);
}
__device__ __forceinline__ v8f wmma_bf16(v16bf a, v16bf b, v8f c) {
    return __builtin_amdgcn_wmma_f32_16x16x32_bf16(false, a, false, b, (short)0, c, false, false);
}

// ---------------- conversions ----------------
__global__ void cvt_bf16(const float* __restrict__ src, unsigned short* __restrict__ dst, int n) {
    int i = blockIdx.x * 256 + threadIdx.x;
    if (i < n) dst[i] = f32_to_bf16(src[i]);
}
// src[K][4096] f32  ->  dst[4096][K] bf16 (transposed, so GEMM B-tiles are contiguous in K)
__global__ void cvt_bf16_T(const float* __restrict__ src, unsigned short* __restrict__ dst, int K) {
    int i = blockIdx.x * 256 + threadIdx.x;
    int n = i & 4095, k = i >> 12;
    if (k < K) dst[(size_t)n * K + k] = f32_to_bf16(src[i]);
}

// ---------------- state init ----------------
__global__ void init_state(float* __restrict__ cbuf, unsigned short* __restrict__ hbuf,
                           float* __restrict__ pooled, float* __restrict__ zpad) {
    int idx = blockIdx.x * 256 + threadIdx.x;            // grid 1024 -> 262144 threads
    cbuf[idx] = 0.0f;
    hbuf[idx] = 0;                                       // bf16 zero
    if (idx < 16 * 1024) pooled[idx] = -__builtin_inff();
    if (idx < 64) zpad[idx] = 0.0f;
}

// ============================================================================
// conv1:  xz[8192][4096] = im2col(x) @ Wx   (3x3, stride 2, pad (0,1))
// Block tile 64(M) x 256(N), BK=32, 8 waves; wave tile 32x64 -> 8 WMMA/chunk.
// Double-buffered LDS fed by async global->LDS b128 copies.
// ============================================================================
__global__ __launch_bounds__(256) void conv1_gemm(const unsigned short* __restrict__ xb,
                                                  const unsigned short* __restrict__ wxT,
                                                  const float* __restrict__ zpad,
                                                  float* __restrict__ xz) {
    __shared__ __attribute__((aligned(32))) unsigned short As[2][64 * 32];   // [m][k], 4KB each
    __shared__ __attribute__((aligned(32))) unsigned short Bs[2][256 * 32];  // [n][k], 16KB each
    const int tid  = threadIdx.x;
    const int lane = tid & 31;
    const int w    = tid >> 5;
    const int wmi  = w >> 2;                 // 0..1 -> M offset 32*wmi
    const int wni  = w & 3;                  // 0..3 -> N offset 64*wni
    const int n0   = blockIdx.x * 256;
    const int m0   = blockIdx.y * 64;

    const unsigned asBase = lds_addr(&As[0][0]);
    const unsigned bsBase = lds_addr(&Bs[0][0]);

    // this thread's fixed staging slots
    const int smi = tid >> 2;                // A: row 0..63
    const int sch = tid & 3;                 // A: 16B chunk (8 bf16)
    const int sm  = m0 + smi;
    const int sbt = sm >> 4, soy = (sm >> 2) & 3, sox = sm & 3;
    const unsigned short* gBrow = wxT + (size_t)(n0 + tid) * 4608;   // B: one n-row per thread

    auto stage = [&](int buf, int kc) {
        const int k0 = kc * 32;
        const int ky = k0 / 1536;
        const int kx = (k0 / 512) % 3;
        const int cb = k0 & 511;
        // A (im2col gather of pre-converted bf16 x; OOB rows -> zero page)
        const int iy = 2 * soy + ky, ix = 2 * sox + kx;
        const void* gA = (iy < 8 && ix < 8)
            ? (const void*)(xb + (((size_t)sbt * 8 + iy) * 8 + ix) * 512 + cb + sch * 8)
            : (const void*)zpad;
        async_b128(asBase + (unsigned)(buf * 4096 + smi * 64 + sch * 16), gA);
        // B: 64 contiguous bytes of Wx^T[n][k0..k0+31]
        const unsigned short* gB = gBrow + k0;
        const unsigned dB = bsBase + (unsigned)(buf * 16384 + tid * 64);
#pragma unroll
        for (int c = 0; c < 4; ++c) async_b128(dB + c * 16, gB + c * 8);
        __builtin_prefetch(gB + 32, 0, 0);   // next chunk of this weight row
    };

    v8f acc[2][4];
#pragma unroll
    for (int i = 0; i < 2; ++i)
#pragma unroll
        for (int j = 0; j < 4; ++j) acc[i][j] = {};

    const int r  = lane & 15;
    const int ka = (lane < 16) ? 0 : 16;
    const int kb = (lane < 16) ? 0 : 16;

    stage(0, 0);
    int buf = 0;
    for (int kc = 0; kc < 144; ++kc) {       // K = 9*512 = 4608
        async_wait0();
        __syncthreads();
        if (kc + 1 < 144) stage(buf ^ 1, kc + 1);   // overlap copy with compute
        const unsigned short* a = &As[buf][0];
        const unsigned short* b = &Bs[buf][0];
        v16bf af[2], bf[4];
#pragma unroll
        for (int i = 0; i < 2; ++i) af[i] = load_a_frag(a, wmi * 32 + i * 16 + r, ka);
#pragma unroll
        for (int j = 0; j < 4; ++j) bf[j] = *(const v16bf*)(b + (wni * 64 + j * 16 + r) * 32 + kb);
#pragma unroll
        for (int i = 0; i < 2; ++i)
#pragma unroll
            for (int j = 0; j < 4; ++j) acc[i][j] = wmma_bf16(af[i], bf[j], acc[i][j]);
        buf ^= 1;
    }
    // epilogue: C/D layout lane->N, vgpr->M
    const int off  = (lane >> 4) << 3;
    const int rowb = m0 + wmi * 32 + off;
    const int colw = n0 + wni * 64 + (lane & 15);
#pragma unroll
    for (int i = 0; i < 2; ++i)
#pragma unroll
        for (int j = 0; j < 4; ++j)
#pragma unroll
            for (int r2 = 0; r2 < 8; ++r2)
                xz[(size_t)(rowb + i * 16 + r2) * 4096 + colw + j * 16] = acc[i][j][r2];
}

// ============================================================================
// conv2 (per step, split-K=4): psum[s][256][4096] = im2col(h) @ Wh (3x3, s1, p1)
// ============================================================================
__global__ __launch_bounds__(256) void conv2_gemm(const unsigned short* __restrict__ hbuf,
                                                  const unsigned short* __restrict__ whT,
                                                  const float* __restrict__ zpad,
                                                  float* __restrict__ psum) {
    __shared__ __attribute__((aligned(32))) unsigned short As[2][64 * 32];
    __shared__ __attribute__((aligned(32))) unsigned short Bs[2][256 * 32];
    const int tid  = threadIdx.x;
    const int lane = tid & 31;
    const int w    = tid >> 5;
    const int wmi  = w >> 2;
    const int wni  = w & 3;
    const int n0   = blockIdx.x * 256;
    const int m0   = blockIdx.y * 64;
    const int sK   = blockIdx.z;             // K-split 0..3, 72 chunks each

    const unsigned asBase = lds_addr(&As[0][0]);
    const unsigned bsBase = lds_addr(&Bs[0][0]);

    const int smi = tid >> 2;
    const int sch = tid & 3;
    const int sm  = m0 + smi;
    const int sbb = sm >> 4, soy = (sm >> 2) & 3, sox = sm & 3;
    const unsigned short* gBrow = whT + (size_t)(n0 + tid) * 9216;

    auto stage = [&](int buf, int kc) {
        const int k0 = kc * 32;
        const int ky = k0 / 3072;
        const int kx = (k0 / 1024) % 3;
        const int cb = k0 & 1023;
        const int iy = soy + ky - 1, ix = sox + kx - 1;
        const void* gA = ((unsigned)iy < 4u && (unsigned)ix < 4u)
            ? (const void*)(hbuf + (((size_t)sbb * 4 + iy) * 4 + ix) * 1024 + cb + sch * 8)
            : (const void*)zpad;
        async_b128(asBase + (unsigned)(buf * 4096 + smi * 64 + sch * 16), gA);
        const unsigned short* gB = gBrow + k0;
        const unsigned dB = bsBase + (unsigned)(buf * 16384 + tid * 64);
#pragma unroll
        for (int c = 0; c < 4; ++c) async_b128(dB + c * 16, gB + c * 8);
        __builtin_prefetch(gB + 32, 0, 0);
    };

    v8f acc[2][4];
#pragma unroll
    for (int i = 0; i < 2; ++i)
#pragma unroll
        for (int j = 0; j < 4; ++j) acc[i][j] = {};

    const int r  = lane & 15;
    const int ka = (lane < 16) ? 0 : 16;
    const int kb = (lane < 16) ? 0 : 16;

    const int kcBeg = sK * 72, kcEnd = kcBeg + 72;       // K = 9*1024 = 9216
    stage(0, kcBeg);
    int buf = 0;
    for (int kc = kcBeg; kc < kcEnd; ++kc) {
        async_wait0();
        __syncthreads();
        if (kc + 1 < kcEnd) stage(buf ^ 1, kc + 1);
        const unsigned short* a = &As[buf][0];
        const unsigned short* b = &Bs[buf][0];
        v16bf af[2], bf[4];
#pragma unroll
        for (int i = 0; i < 2; ++i) af[i] = load_a_frag(a, wmi * 32 + i * 16 + r, ka);
#pragma unroll
        for (int j = 0; j < 4; ++j) bf[j] = *(const v16bf*)(b + (wni * 64 + j * 16 + r) * 32 + kb);
#pragma unroll
        for (int i = 0; i < 2; ++i)
#pragma unroll
            for (int j = 0; j < 4; ++j) acc[i][j] = wmma_bf16(af[i], bf[j], acc[i][j]);
        buf ^= 1;
    }
    const int off  = (lane >> 4) << 3;
    const int rowb = m0 + wmi * 32 + off;
    const int colw = n0 + wni * 64 + (lane & 15);
    float* ps = psum + (size_t)sK * 256 * 4096;
#pragma unroll
    for (int i = 0; i < 2; ++i)
#pragma unroll
        for (int j = 0; j < 4; ++j)
#pragma unroll
            for (int r2 = 0; r2 < 8; ++r2)
                ps[(size_t)(rowb + i * 16 + r2) * 4096 + colw + j * 16] = acc[i][j][r2];
}

// ---------------- LSTM gates + state update + running global max-pool ----------------
__global__ void lstm_pointwise(const float* __restrict__ xz, const float* __restrict__ psum,
                               float* __restrict__ cbuf, unsigned short* __restrict__ hbuf,
                               float* __restrict__ pooled, int t) {
    const int idx = blockIdx.x * 256 + threadIdx.x;      // (b,oy,ox)*1024 + f
    const int m = idx >> 10, f = idx & 1023;
    const int bi = m >> 4, sp = m & 15;
    const float* xzr = xz + (size_t)(bi * 512 + t * 16 + sp) * 4096;
    float g[4];
#pragma unroll
    for (int gate = 0; gate < 4; ++gate) {
        float v = xzr[gate * 1024 + f];
#pragma unroll
        for (int s = 0; s < 4; ++s)
            v += psum[((size_t)s * 256 + m) * 4096 + gate * 1024 + f];
        g[gate] = v;
    }
    const float ig = hsig(g[0]), fg = hsig(g[1]);
    const float cn = fg * cbuf[idx] + ig * tanhf(g[2]);
    const float hn = hsig(g[3]) * tanhf(cn);
    cbuf[idx] = cn;
    hbuf[idx] = f32_to_bf16(hn);
    atomicMaxF(pooled + ((m >> 4) << 10) + f, hn);
}

// ---------------- dense (1024x2) + softmax ----------------
__global__ __launch_bounds__(256) void dense_softmax(const float* __restrict__ pooled,
                                                     const float* __restrict__ Wd,
                                                     const float* __restrict__ bd,
                                                     float* __restrict__ out) {
    const int b = blockIdx.x, tid = threadIdx.x;
    __shared__ float s0[256], s1[256];
    float a0 = 0.0f, a1 = 0.0f;
    for (int f = tid; f < 1024; f += 256) {
        const float p = pooled[b * 1024 + f];
        a0 += p * Wd[2 * f];
        a1 += p * Wd[2 * f + 1];
    }
    s0[tid] = a0; s1[tid] = a1;
    __syncthreads();
    for (int s = 128; s > 0; s >>= 1) {
        if (tid < s) { s0[tid] += s0[tid + s]; s1[tid] += s1[tid + s]; }
        __syncthreads();
    }
    if (tid == 0) {
        const float l0 = s0[0] + bd[0], l1 = s1[0] + bd[1];
        const float mx = fmaxf(l0, l1);
        const float e0 = __expf(l0 - mx), e1 = __expf(l1 - mx);
        const float inv = 1.0f / (e0 + e1);
        out[b * 2 + 0] = e0 * inv;
        out[b * 2 + 1] = e1 * inv;
    }
}

// ---------------- host ----------------
extern "C" void kernel_launch(void* const* d_in, const int* in_sizes, int n_in,
                              void* d_out, int out_size, void* d_ws, size_t ws_size,
                              hipStream_t stream) {
    const float* x  = (const float*)d_in[0];
    const float* Wx = (const float*)d_in[1];
    const float* Wh = (const float*)d_in[2];
    const float* Wd = (const float*)d_in[3];
    const float* bd = (const float*)d_in[4];
    float* out = (float*)d_out;

    char* ws = (char*)d_ws;
    unsigned short* wxT   = (unsigned short*)(ws);                 //  37,748,736 B  [4096][4608]
    unsigned short* whT   = (unsigned short*)(ws + 37748736);      //  75,497,472 B  [4096][9216]
    unsigned short* xb    = (unsigned short*)(ws + 113246208);     //  33,554,432 B  bf16 x
    float*          xz    = (float*)(ws + 146800640);              // 134,217,728 B
    float*          psum  = (float*)(ws + 281018368);              //  16,777,216 B  [4][256][4096]
    float*          cbuf  = (float*)(ws + 297795584);              //   1,048,576 B
    unsigned short* hbuf  = (unsigned short*)(ws + 298844160);     //     524,288 B
    float*          pooled= (float*)(ws + 299368448);              //      65,536 B
    float*          zpad  = (float*)(ws + 299433984);              //         256 B

    const int XN = 16 * 32 * 8 * 8 * 512;                          // 16,777,216
    cvt_bf16_T<<<(4608 * 4096) / 256, 256, 0, stream>>>(Wx, wxT, 4608);
    cvt_bf16_T<<<(9216 * 4096) / 256, 256, 0, stream>>>(Wh, whT, 9216);
    cvt_bf16<<<XN / 256, 256, 0, stream>>>(x, xb, XN);
    init_state<<<1024, 256, 0, stream>>>(cbuf, hbuf, pooled, zpad);

    // xz: M=8192 (128 tiles of 64), N=4096 (16 tiles of 256)
    conv1_gemm<<<dim3(16, 128), 256, 0, stream>>>(xb, wxT, zpad, xz);

    for (int t = 0; t < 32; ++t) {
        conv2_gemm<<<dim3(16, 4, 4), 256, 0, stream>>>(hbuf, whT, zpad, psum);
        lstm_pointwise<<<1024, 256, 0, stream>>>(xz, psum, cbuf, hbuf, pooled, t);
    }

    dense_softmax<<<16, 256, 0, stream>>>(pooled, Wd, bd, out);
}
